// EndpointVectorField_14869176779181
// MI455X (gfx1250) — matrix operand order repaired and checked by reading, compiled, not run on hardware
//
#include <hip/hip_runtime.h>
#include <hip/hip_bf16.h>

// ---------------------------------------------------------------------------
// MI455X (gfx1250) implementation.
// EdgeUpdate: wave-per-16-edge-tile GEMM on v_wmma_f32_16x16x32_f16
//   x[16,224(pad)] @ W1[224,64] -> silu -> @ W2[64,64] -> silu -> +res -> LN
//   W1 B-fragments pre-swizzled in LDS (batched ds_load_b128 per K-chunk),
//   W2 B-fragments held in VGPRs for the whole loop, residual loads issued
//   ahead of layer-2 WMMAs.
// NodePositionUpdate: wave-per-node VALU GVP chain (negligible cost).
// ---------------------------------------------------------------------------

typedef _Float16 v16h __attribute__((ext_vector_type(16)));
typedef float    v8f  __attribute__((ext_vector_type(8)));

union FragH { v16h h; unsigned u[8]; uint4 q[2]; };

__device__ __forceinline__ unsigned pk2(float a, float b) {
  union { _Float16 h[2]; unsigned u; } x;
  x.h[0] = (_Float16)a; x.h[1] = (_Float16)b;
  return x.u;
}
__device__ __forceinline__ float silu_f(float x) { return x / (1.f + __expf(-x)); }
__device__ __forceinline__ float sigm_f(float x) { return 1.f / (1.f + __expf(-x)); }

// Build a 16x32 f16 A-fragment for this lane from a contiguous f32 row.
// Lane l (l<16): K = koff..koff+7 (VGPR0-3), koff+16..koff+23 (VGPR4-7), koff=0.
// Lane l>=16: koff = 8.  (CDNA5 ISA 7.12.2 16-bit A layout.)
__device__ __forceinline__ v16h load_a_chunk(const float* base, int koff) {
  const float4* p = (const float4*)(base + koff);
  float4 a0 = p[0], a1 = p[1];
  const float4* q = (const float4*)(base + koff + 16);
  float4 b0 = q[0], b1 = q[1];
  FragH f;
  f.u[0] = pk2(a0.x, a0.y); f.u[1] = pk2(a0.z, a0.w);
  f.u[2] = pk2(a1.x, a1.y); f.u[3] = pk2(a1.z, a1.w);
  f.u[4] = pk2(b0.x, b0.y); f.u[5] = pk2(b0.z, b0.w);
  f.u[6] = pk2(b1.x, b1.y); f.u[7] = pk2(b1.z, b1.w);
  return f.h;
}

// One K-chunk vs 4 N-tiles: load ALL 4 B fragments first (8 ds_load_b128
// issued as a batch so one dscnt wait covers them), then 4 WMMAs.
__device__ __forceinline__ void mm4(v8f acc[4], v16h a, const unsigned* fragc, int lane) {
  FragH b[4];
#pragma unroll
  for (int t = 0; t < 4; ++t) {
    const uint4* p = (const uint4*)(fragc + t * 256 + lane * 8);
    b[t].q[0] = p[0];
    b[t].q[1] = p[1];
  }
#pragma unroll
  for (int t = 0; t < 4; ++t)
    acc[t] = __builtin_amdgcn_wmma_f32_16x16x32_f16(
        false, a, false, b[t].h, (short)0, acc[t], false, false);
}

#define WPB 8  // waves per block

__global__ __launch_bounds__(256) void edge_update_kernel(
    const float* __restrict__ ns, const float* __restrict__ efeat,
    const float* __restrict__ dd, const int* __restrict__ src,
    const int* __restrict__ dst, const float* __restrict__ W1,
    const float* __restrict__ b1, const float* __restrict__ W2,
    const float* __restrict__ b2, const float* __restrict__ lng,
    const float* __restrict__ lnb, float* __restrict__ out, int E) {
  // B-fragment banks: [frag][lane][8 dwords], frag = kchunk*4 + ntile
  __shared__ unsigned w1frag[28 * 256];  // 28 KB (K padded 208 -> 224)
  __shared__ unsigned w2frag[8 * 256];   //  8 KB (read once into VGPRs)
  __shared__ __align__(16) _Float16 hbuf[WPB][16 * 64];  // 16 KB transpose buf

  const int tid  = threadIdx.x;
  const int lane = tid & 31;
  const int wave = tid >> 5;

  // ---- stage W1/W2 as f16 B fragments (B layout: lanes0-15 N=lane K=0..7,16..23,
  //      lanes16-31 N=lane-16 K=8..15,24..31; matches A table order) ----
  for (int idx = tid; idx < 28 * 256; idx += 256) {
    int f = idx >> 8, rem = idx & 255;
    int l = rem >> 3, i = rem & 7;
    int c = f >> 2, t = f & 3;
    int k = 32 * c + ((l < 16) ? 0 : 8) + ((i < 4) ? 0 : 8) + 2 * i;
    int n = 16 * t + (l & 15);
    float va = (k     < 208) ? W1[k * 64 + n]       : 0.f;
    float vb = (k + 1 < 208) ? W1[(k + 1) * 64 + n] : 0.f;
    w1frag[f * 256 + l * 8 + i] = pk2(va, vb);
  }
  for (int idx = tid; idx < 8 * 256; idx += 256) {
    int f = idx >> 8, rem = idx & 255;
    int l = rem >> 3, i = rem & 7;
    int c = f >> 2, t = f & 3;
    int k = 32 * c + ((l < 16) ? 0 : 8) + ((i < 4) ? 0 : 8) + 2 * i;
    int n = 16 * t + (l & 15);
    w2frag[f * 256 + l * 8 + i] = pk2(W2[k * 64 + n], W2[(k + 1) * 64 + n]);
  }
  __syncthreads();

  const int l15  = lane & 15;
  const int hi   = lane >> 4;  // half-wave id
  const int koff = hi * 8;
  const int hi8  = hi * 8;

  // W2 fragments live in VGPRs for the whole loop (8 frags = 64 VGPRs).
  FragH w2r[8];
#pragma unroll
  for (int f = 0; f < 8; ++f) {
    const uint4* p = (const uint4*)(w2frag + f * 256 + lane * 8);
    w2r[f].q[0] = p[0];
    w2r[f].q[1] = p[1];
  }

  float b1v[4], b2v[4], gv[4], bv[4];
#pragma unroll
  for (int t = 0; t < 4; ++t) {
    b1v[t] = b1[16 * t + l15];
    b2v[t] = b2[16 * t + l15];
    gv[t]  = lng[16 * t + l15];
    bv[t]  = lnb[16 * t + l15];
  }

  _Float16* hb = hbuf[wave];
  const int tiles = (E + 15) >> 4;
  const int gw = blockIdx.x * WPB + wave;
  const int nw = gridDim.x * WPB;

  for (int tile = gw; tile < tiles; tile += nw) {
    const int e0 = tile * 16;
    int e = e0 + l15;
    int ec = (e < E) ? e : (E - 1);
    const int is = src[ec];
    const int id = dst[ec];
    const float* rs = ns + (size_t)is * 64;
    const float* rd = ns + (size_t)id * 64;
    const float* re = efeat + (size_t)ec * 64;
    const float* rr = dd + (size_t)ec * 16;

    // ---------------- layer 1: x[16,224] @ W1 ----------------
    v8f acc[4];
#pragma unroll
    for (int t = 0; t < 4; ++t)
#pragma unroll
      for (int j = 0; j < 8; ++j) acc[t][j] = b1v[t];

    mm4(acc, load_a_chunk(rs, koff),      w1frag + 0 * 1024, lane);   // src 0..31
    mm4(acc, load_a_chunk(rs + 32, koff), w1frag + 1 * 1024, lane);   // src 32..63
    mm4(acc, load_a_chunk(rd, koff),      w1frag + 2 * 1024, lane);   // dst 0..31
    mm4(acc, load_a_chunk(rd + 32, koff), w1frag + 3 * 1024, lane);   // dst 32..63
    mm4(acc, load_a_chunk(re, koff),      w1frag + 4 * 1024, lane);   // ef 0..31
    mm4(acc, load_a_chunk(re + 32, koff), w1frag + 5 * 1024, lane);   // ef 32..63
    {
      // chunk 6: d[0..15] then zero pad (cols 208..223)
      const float4* p = (const float4*)(rr + koff);
      float4 a0 = p[0], a1 = p[1];
      FragH f;
      f.u[0] = pk2(a0.x, a0.y); f.u[1] = pk2(a0.z, a0.w);
      f.u[2] = pk2(a1.x, a1.y); f.u[3] = pk2(a1.z, a1.w);
      f.u[4] = 0u; f.u[5] = 0u; f.u[6] = 0u; f.u[7] = 0u;
      mm4(acc, f.h, w1frag + 6 * 1024, lane);
    }

    // ---------------- silu + C->A transpose through LDS ----------------
#pragma unroll
    for (int t = 0; t < 4; ++t)
#pragma unroll
      for (int j = 0; j < 8; ++j) {
        int M = j + hi8;
        int k = 16 * t + l15;
        hb[M * 64 + k] = (_Float16)silu_f(acc[t][j]);
      }
    __asm__ volatile("" ::: "memory");  // keep ds stores before ds loads

    // Residual loads issued early so they overlap the layer-2 WMMAs.
    float fe[4][8];
#pragma unroll
    for (int t = 0; t < 4; ++t)
#pragma unroll
      for (int j = 0; j < 8; ++j) {
        int ee = e0 + j + hi8;
        fe[t][j] = (ee < E) ? efeat[(size_t)ee * 64 + 16 * t + l15] : 0.f;
      }

    // ---------------- layer 2: h[16,64] @ W2 (B in VGPRs) ----------------
    v8f acc2[4];
#pragma unroll
    for (int t = 0; t < 4; ++t)
#pragma unroll
      for (int j = 0; j < 8; ++j) acc2[t][j] = b2v[t];

#pragma unroll
    for (int c2 = 0; c2 < 2; ++c2) {
      FragH a;
      const _Float16* hp = hb + l15 * 64 + 32 * c2 + koff;
      a.q[0] = *(const uint4*)(hp);
      a.q[1] = *(const uint4*)(hp + 16);
#pragma unroll
      for (int t = 0; t < 4; ++t)
        acc2[t] = __builtin_amdgcn_wmma_f32_16x16x32_f16(
            false, a.h, false, w2r[c2 * 4 + t].h, (short)0, acc2[t], false,
            false);
    }
    __asm__ volatile("" ::: "memory");

    // ---------------- silu + residual + LayerNorm + store ----------------
    float y[4][8];
#pragma unroll
    for (int t = 0; t < 4; ++t)
#pragma unroll
      for (int j = 0; j < 8; ++j) y[t][j] = fe[t][j] + silu_f(acc2[t][j]);

#pragma unroll
    for (int j = 0; j < 8; ++j) {
      float s1 = y[0][j] + y[1][j] + y[2][j] + y[3][j];
      float s2 = y[0][j] * y[0][j] + y[1][j] * y[1][j] +
                 y[2][j] * y[2][j] + y[3][j] * y[3][j];
#pragma unroll
      for (int m = 1; m <= 8; m <<= 1) {  // reduce within 16-lane half (a row)
        s1 += __shfl_xor(s1, m, 32);
        s2 += __shfl_xor(s2, m, 32);
      }
      float mean = s1 * (1.f / 64.f);
      float var  = s2 * (1.f / 64.f) - mean * mean;
      float rstd = rsqrtf(var + 1e-5f);
      int ee = e0 + j + hi8;
      if (ee < E) {
#pragma unroll
        for (int t = 0; t < 4; ++t)
          out[(size_t)ee * 64 + 16 * t + l15] =
              (y[t][j] - mean) * rstd * gv[t] + bv[t];
      }
    }
  }
}

// ---------------------------------------------------------------------------
// Node GVP chain: one wave per node, small per-wave LDS scratch.
// ---------------------------------------------------------------------------
__global__ __launch_bounds__(256) void node_gvp_kernel(
    const float* __restrict__ ns, const float* __restrict__ pos,
    const float* __restrict__ vec,
    const float* Wh0, const float* Ws0, const float* bs0, const float* Wu0,
    const float* Wg0, const float* bg0,
    const float* Wh1, const float* Ws1, const float* bs1, const float* Wu1,
    const float* Wg1, const float* bg1,
    const float* Wh2, const float* Ws2, const float* bs2, const float* Wu2,
    const float* Wg2, const float* bg2,
    float* __restrict__ out_pos, int N) {
  __shared__ float sc[WPB][256];
  const int lane = threadIdx.x & 31;
  const int wave = threadIdx.x >> 5;
  const int n = blockIdx.x * WPB + wave;
  if (n >= N) return;

  float* S  = sc[wave];        // s        [64]
  float* S2 = S + 64;          // s_out    [64]
  float* VN = S + 128;         // vn       [16]
  float* VH = S + 144;         // Vh       [16*3]
  float* VV = S + 192;         // V        [16*3]

  S[lane]      = ns[(size_t)n * 64 + lane];
  S[lane + 32] = ns[(size_t)n * 64 + 32 + lane];
  VV[lane] = vec[(size_t)n * 48 + lane];
  if (lane < 16) VV[lane + 32] = vec[(size_t)n * 48 + 32 + lane];
  __asm__ volatile("" ::: "memory");

  const float* Wh[3] = {Wh0, Wh1, Wh2};
  const float* Ws[3] = {Ws0, Ws1, Ws2};
  const float* bs[3] = {bs0, bs1, bs2};
  const float* Wu[3] = {Wu0, Wu1, Wu2};
  const float* Wg[3] = {Wg0, Wg1, Wg2};
  const float* bg[3] = {bg0, bg1, bg2};
  const int vo[3] = {16, 16, 1};

#pragma unroll
  for (int g = 0; g < 3; ++g) {
    // Vh[h][c] = sum_v V[v][c] * Wh[v][h];  vn[h] = |Vh[h]|
    if (lane < 16) {
      float a0 = 0.f, a1 = 0.f, a2 = 0.f;
#pragma unroll
      for (int v = 0; v < 16; ++v) {
        float w = Wh[g][v * 16 + lane];
        a0 += VV[v * 3 + 0] * w;
        a1 += VV[v * 3 + 1] * w;
        a2 += VV[v * 3 + 2] * w;
      }
      VH[lane * 3 + 0] = a0;
      VH[lane * 3 + 1] = a1;
      VH[lane * 3 + 2] = a2;
      VN[lane] = __fsqrt_rn(a0 * a0 + a1 * a1 + a2 * a2 + 1e-8f);
    }
    __asm__ volatile("" ::: "memory");

    // s_out = silu([s, vn] @ Ws + bs); each lane owns 2 output columns
    {
      int j0 = lane, j1 = lane + 32;
      float a0 = bs[g][j0], a1 = bs[g][j1];
      for (int k = 0; k < 64; ++k) {
        float s = S[k];
        a0 += s * Ws[g][k * 64 + j0];
        a1 += s * Ws[g][k * 64 + j1];
      }
#pragma unroll
      for (int k = 0; k < 16; ++k) {
        float s = VN[k];
        a0 += s * Ws[g][(64 + k) * 64 + j0];
        a1 += s * Ws[g][(64 + k) * 64 + j1];
      }
      S2[j0] = silu_f(a0);
      S2[j1] = silu_f(a1);
    }
    __asm__ volatile("" ::: "memory");

    // gate = act(s_out @ Wg + bg); Vnew[u] = (Wu^T Vh)[u] * gate[u]
    if (lane < vo[g]) {
      float gi = bg[g][lane];
      for (int j = 0; j < 64; ++j) gi += S2[j] * Wg[g][j * vo[g] + lane];
      float gate = (g < 2) ? sigm_f(gi) : gi;
      float u0 = 0.f, u1 = 0.f, u2 = 0.f;
#pragma unroll
      for (int h = 0; h < 16; ++h) {
        float w = Wu[g][h * vo[g] + lane];
        u0 += VH[h * 3 + 0] * w;
        u1 += VH[h * 3 + 1] * w;
        u2 += VH[h * 3 + 2] * w;
      }
      if (g < 2) {
        VV[lane * 3 + 0] = u0 * gate;
        VV[lane * 3 + 1] = u1 * gate;
        VV[lane * 3 + 2] = u2 * gate;
      } else if (lane == 0) {
        out_pos[(size_t)n * 3 + 0] = pos[(size_t)n * 3 + 0] + u0 * gate;
        out_pos[(size_t)n * 3 + 1] = pos[(size_t)n * 3 + 1] + u1 * gate;
        out_pos[(size_t)n * 3 + 2] = pos[(size_t)n * 3 + 2] + u2 * gate;
      }
    }
    // s <- s_out
    S[lane] = S2[lane];
    S[lane + 32] = S2[lane + 32];
    __asm__ volatile("" ::: "memory");
  }
}

extern "C" void kernel_launch(void* const* d_in, const int* in_sizes, int n_in,
                              void* d_out, int out_size, void* d_ws,
                              size_t ws_size, hipStream_t stream) {
  const float* ns    = (const float*)d_in[0];
  const float* efeat = (const float*)d_in[1];
  const float* dd    = (const float*)d_in[2];
  const int*   src   = (const int*)d_in[3];
  const int*   dst   = (const int*)d_in[4];
  const float* pos   = (const float*)d_in[5];
  const float* vec   = (const float*)d_in[6];
  const float* W1    = (const float*)d_in[7];
  const float* b1    = (const float*)d_in[8];
  const float* W2    = (const float*)d_in[9];
  const float* b2    = (const float*)d_in[10];
  const float* lng   = (const float*)d_in[11];
  const float* lnb   = (const float*)d_in[12];

  const int E = in_sizes[3];
  const int N = in_sizes[0] / 64;

  float* out_edge = (float*)d_out;
  float* out_pos  = out_edge + (size_t)E * 64;

  edge_update_kernel<<<800, 256, 0, stream>>>(ns, efeat, dd, src, dst, W1, b1,
                                              W2, b2, lng, lnb, out_edge, E);

  node_gvp_kernel<<<(N + WPB - 1) / WPB, 256, 0, stream>>>(
      ns, pos, vec,
      (const float*)d_in[13], (const float*)d_in[14], (const float*)d_in[15],
      (const float*)d_in[16], (const float*)d_in[17], (const float*)d_in[18],
      (const float*)d_in[19], (const float*)d_in[20], (const float*)d_in[21],
      (const float*)d_in[22], (const float*)d_in[23], (const float*)d_in[24],
      (const float*)d_in[25], (const float*)d_in[26], (const float*)d_in[27],
      (const float*)d_in[28], (const float*)d_in[29], (const float*)d_in[30],
      out_pos, N);
}
